// ACEGCNClassifier_16363825398627
// MI455X (gfx1250) — compile-verified, matrix operand order checked by reading, not run on hardware
//
#include <hip/hip_runtime.h>
#include <math.h>

// Problem constants (match reference)
static constexpr int kB  = 16;
static constexpr int kL  = 512;
static constexpr int kD  = 768;
static constexpr int kH  = 4;
static constexpr int kA  = 100;   // true feature dim
static constexpr int kAP = 128;   // padded feature dim (multiple of 32 for WMMA K, 16 for N)
static constexpr int kP  = 3;
static constexpr int kNL = 2;
static constexpr int kM  = kB * kL;         // 8192 token rows
static constexpr int kWx = kH + 2 * kA;     // 204, row stride of Wx_w

typedef __attribute__((ext_vector_type(16))) _Float16 v16h;
typedef __attribute__((ext_vector_type(8)))  _Float16 v8h;
typedef __attribute__((ext_vector_type(8)))  float    v8f;

// ---------------------------------------------------------------------------
// Wave-per-tile WMMA GEMM, 16(M) x 32(N) output per wave, K contiguous:
//   C[m,n] = sum_k A[m,k]*B[n,k] (+bias, relu)
// A element (m,k) at A + m*sAm + k ; B element (n,k) at B + n*sBn + k.
// Requirements: K % 32 == 0; sAm, sBn multiples of 8; 16B-aligned bases.
// grid.x = M/16, grid.y = N/32, grid.z = batch; blockDim.x = 32 (one wave).
// v16h lane layout (CDNA5 16-bit A/B tiles): lane&15 selects m (or n);
// lanes 0-15 hold K = {kb+0..7, kb+16..23}, lanes 16-31 hold K = {kb+8..15, kb+24..31},
// i.e. two aligned 8-half (128-bit) chunks per lane per fragment.
// ---------------------------------------------------------------------------
__device__ inline v16h frag16(const _Float16* p, long k0)
{
    const v8h lo = *(const v8h*)(p + k0);
    const v8h hi = *(const v8h*)(p + k0 + 16);
    return __builtin_shufflevector(lo, hi, 0, 1, 2, 3, 4, 5, 6, 7,
                                           8, 9, 10, 11, 12, 13, 14, 15);
}

__global__ void k_wmma_gemm(const _Float16* __restrict__ A, long sAm, long sAb,
                            const _Float16* __restrict__ Bp, long sBn, long sBb,
                            float* __restrict__ C, long ldc, long sCb,
                            int K, const float* __restrict__ bias, int biasN, int doRelu)
{
    const int lane = threadIdx.x;                 // 32 threads = 1 wave
    const long m0 = (long)blockIdx.x * 16;
    const long n0 = (long)blockIdx.y * 32;
    const _Float16* Ab = A  + (long)blockIdx.z * sAb;
    const _Float16* Bb = Bp + (long)blockIdx.z * sBb;
    float*          Cb = C  + (long)blockIdx.z * sCb;

    const int  lo16 = lane & 15;
    const int  kh   = (lane >> 4) << 3;           // 0 or 8
    const _Float16* ap  = Ab + (m0 + lo16) * sAm;
    const _Float16* bp0 = Bb + (n0 + lo16) * sBn;
    const _Float16* bp1 = bp0 + 16 * sBn;

    v8f acc0 = {};
    v8f acc1 = {};
#pragma unroll 2
    for (int kb = 0; kb < K; kb += 32) {
        const long k0 = kb + kh;
        // prefetch next K-step fragments (speculative; safe past end)
        __builtin_prefetch(ap  + k0 + 32, 0, 1);
        __builtin_prefetch(bp0 + k0 + 32, 0, 1);
        __builtin_prefetch(bp1 + k0 + 32, 0, 1);
        const v16h a  = frag16(ap,  k0);
        const v16h b0 = frag16(bp0, k0);
        const v16h b1 = frag16(bp1, k0);
        acc0 = __builtin_amdgcn_wmma_f32_16x16x32_f16(false, a, false, b0,
                                                      (short)0, acc0, false, false);
        acc1 = __builtin_amdgcn_wmma_f32_16x16x32_f16(false, a, false, b1,
                                                      (short)0, acc1, false, false);
    }

    const int n0i = (int)(n0 + lo16);
    const int n1i = n0i + 16;
    const float bv0 = (bias != nullptr && n0i < biasN) ? bias[n0i] : 0.0f;
    const float bv1 = (bias != nullptr && n1i < biasN) ? bias[n1i] : 0.0f;
#pragma unroll
    for (int v = 0; v < 8; ++v) {
        const long m = m0 + v + kh;               // C/D layout: m = v + 8*(lane>=16)
        float r0 = acc0[v] + bv0;
        float r1 = acc1[v] + bv1;
        if (doRelu) { r0 = fmaxf(r0, 0.0f); r1 = fmaxf(r1, 0.0f); }
        Cb[m * ldc + n0i] = r0;
        Cb[m * ldc + n1i] = r1;
    }
}

// ---------------------------------------------------------------------------
// Block reductions (blockDim.x <= 256)
// ---------------------------------------------------------------------------
__device__ inline float block_sum(float v, float* sm) {
    int t = threadIdx.x;
    sm[t] = v; __syncthreads();
    for (int s = blockDim.x >> 1; s > 0; s >>= 1) {
        if (t < s) sm[t] += sm[t + s];
        __syncthreads();
    }
    float r = sm[0]; __syncthreads();
    return r;
}
__device__ inline float block_max(float v, float* sm) {
    int t = threadIdx.x;
    sm[t] = v; __syncthreads();
    for (int s = blockDim.x >> 1; s > 0; s >>= 1) {
        if (t < s) sm[t] = fmaxf(sm[t], sm[t + s]);
        __syncthreads();
    }
    float r = sm[0]; __syncthreads();
    return r;
}

// LayerNorm (ddof=1) -> f16 rows. One block per token row.
__global__ void k_layernorm(const float* __restrict__ seq, const float* __restrict__ ga,
                            const float* __restrict__ be, _Float16* __restrict__ outp)
{
    __shared__ float sm[256];
    const long row = blockIdx.x;
    const float* p = seq + row * (long)kD;
    float s = 0.0f;
    for (int i = threadIdx.x; i < kD; i += blockDim.x) s += p[i];
    const float mean = block_sum(s, sm) / (float)kD;
    float vs = 0.0f;
    for (int i = threadIdx.x; i < kD; i += blockDim.x) { float d = p[i] - mean; vs += d * d; }
    const float stdv = sqrtf(block_sum(vs, sm) / (float)(kD - 1));
    const float inv  = 1.0f / (stdv + 1e-6f);
    for (int i = threadIdx.x; i < kD; i += blockDim.x)
        outp[row * (long)kD + i] = (_Float16)(ga[i] * (p[i] - mean) * inv + be[i]);
}

// Pad/convert an f32 weight block into a zero-padded f16 matrix region.
__global__ void k_pack_weight(const float* __restrict__ w, int rows, int cols, int ldw, int colOff,
                              _Float16* __restrict__ outp, long outLd, int outColOff,
                              int outRows, int outColsBlock)
{
    const long idx = (long)blockIdx.x * blockDim.x + threadIdx.x;
    if (idx >= (long)outRows * outColsBlock) return;
    const int r = (int)(idx / outColsBlock);
    const int c = (int)(idx % outColsBlock);
    const float v = (r < rows && c < cols) ? w[(long)r * ldw + colOff + c] : 0.0f;
    outp[(long)r * outLd + outColOff + c] = (_Float16)v;
}

// Strided f32 -> f16 copy (rows x cols), for stuffing xcat slots / axh.
__global__ void k_f32_to_f16(const float* __restrict__ in, long ldi,
                             _Float16* __restrict__ outp, long ldo, long colOff,
                             long rows, int cols)
{
    const long idx = (long)blockIdx.x * blockDim.x + threadIdx.x;
    if (idx >= rows * cols) return;
    const long r = idx / cols;
    const long c = idx % cols;
    outp[r * ldo + colOff + c] = (_Float16)in[r * ldi + c];
}

// d-major transpose of x_g for the aggregation GEMM's B operand:
// xT[b, d, j] = x_g[(b*kL + j)*kAP + d]   (f16 out, K=j contiguous)
__global__ void k_transpose_x(const float* __restrict__ x, _Float16* __restrict__ xT)
{
    const long idx = (long)blockIdx.x * blockDim.x + threadIdx.x;
    if (idx >= (long)kB * kAP * kL) return;
    const long j = idx % kL;
    const long d = (idx / kL) % kAP;
    const long b = idx / ((long)kAP * kL);
    xT[idx] = (_Float16)x[(b * kL + j) * (long)kAP + d];
}

// Scatter q/k (B*L, AP) -> per-head padded (B,H,L,32) f16; 1/sqrt(25) folded into q.
__global__ void k_pack_qk(const float* __restrict__ q32, const float* __restrict__ k32,
                          _Float16* __restrict__ qp, _Float16* __restrict__ kp)
{
    const long idx = (long)blockIdx.x * blockDim.x + threadIdx.x;
    if (idx >= (long)kB * kH * kL * 32) return;
    const int  d = (int)(idx & 31);
    const long t = idx >> 5;                 // b*H*L + h*L + l
    const long l = t % kL;
    const long h = (t / kL) % kH;
    const long b = t / ((long)kL * kH);
    float qv = 0.0f, kv = 0.0f;
    if (d < 25) {
        const long src = (b * kL + l) * (long)kAP + h * 25 + d;
        qv = q32[src] * 0.2f;                // 1/sqrt(DK)
        kv = k32[src];
    }
    qp[idx] = (_Float16)qv;
    kp[idx] = (_Float16)kv;
}

// In-place masked softmax over j with syntax bias. One block per (b,h,i) row.
__global__ void k_softmax(float* __restrict__ adj, const float* __restrict__ syn,
                          const int* __restrict__ srcm)
{
    __shared__ float sm[256];
    const long row = blockIdx.x;             // (b*H + h)*L + i
    const long b   = row / ((long)kH * kL);
    float*       p  = adj + row * (long)kL;
    const float* sy = syn + row * (long)kL;
    const int*   mk = srcm + b * (long)kL;
    float mx = -3.4e38f;
    for (int j = threadIdx.x; j < kL; j += blockDim.x) {
        const float v = (mk[j] == 0 ? -1.0e9f : p[j]) + sy[j];
        p[j] = v;
        mx = fmaxf(mx, v);
    }
    mx = block_max(mx, sm);
    float s = 0.0f;
    for (int j = threadIdx.x; j < kL; j += blockDim.x) {
        const float e = expf(p[j] - mx);
        p[j] = e; s += e;
    }
    const float inv = 1.0f / block_sum(s, sm);
    for (int j = threadIdx.x; j < kL; j += blockDim.x) p[j] *= inv;
}

// adjsum[b,i,j] = (1/H) * sum_h adj[b,h,i,j]  (f16 out, feeds the Ax GEMM)
__global__ void k_adjsum(const float* __restrict__ adj, _Float16* __restrict__ outp)
{
    const long idx = (long)blockIdx.x * blockDim.x + threadIdx.x;
    if (idx >= (long)kB * kL * kL) return;
    const long b   = idx / ((long)kL * kL);
    const long rem = idx % ((long)kL * kL);
    const float* base = adj + b * (long)kH * kL * kL + rem;
    float s = 0.0f;
#pragma unroll
    for (int h = 0; h < kH; ++h) s += base[(long)h * kL * kL];
    outp[idx] = (_Float16)(s * 0.25f);
}

// u[b,g,j] = x_g[b,j,:]·B1[g,:] ; v[b,g,i] = x_g[b,i,:]·B2[g,:]
__global__ void k_uv(const float* __restrict__ xg, const float* __restrict__ Wx_w,
                     float* __restrict__ u, float* __restrict__ v)
{
    const long idx = (long)blockIdx.x * blockDim.x + threadIdx.x;
    if (idx >= (long)kB * kH * kL) return;
    const long l = idx % kL;
    const long g = (idx / kL) % kH;
    const long b = idx / ((long)kH * kL);
    const float* xr = xg + (b * kL + l) * (long)kAP;
    const float* b1 = Wx_w + g * (long)kWx + kH;
    const float* b2 = b1 + kA;
    float su = 0.0f, sv = 0.0f;
    for (int d = 0; d < kA; ++d) { su += xr[d] * b1[d]; sv += xr[d] * b2[d]; }
    u[idx] = su; v[idx] = sv;
}

// adj[b,g,i,j] = sum_h Aw[g,h]*adj_old[b,h,i,j] + u[b,g,j] + v[b,g,i] + Wx_b[g]
// In-place safe: each thread owns one (b,i,j) across all h/g.
__global__ void k_adj_update(float* __restrict__ adj, const float* __restrict__ u,
                             const float* __restrict__ v, const float* __restrict__ Wx_w,
                             const float* __restrict__ Wx_b)
{
    const long idx = (long)blockIdx.x * blockDim.x + threadIdx.x;
    if (idx >= (long)kB * kL * kL) return;
    const long b   = idx / ((long)kL * kL);
    const long rem = idx % ((long)kL * kL);
    const long i   = rem / kL;
    const long j   = rem % kL;
    float* base = adj + b * (long)kH * kL * kL + rem;
    float av[kH];
#pragma unroll
    for (int h = 0; h < kH; ++h) av[h] = base[(long)h * kL * kL];
    float nv[kH];
#pragma unroll
    for (int g = 0; g < kH; ++g) {
        float s = Wx_b[g];
#pragma unroll
        for (int h = 0; h < kH; ++h) s += Wx_w[(long)g * kWx + h] * av[h];
        s += u[(b * kH + g) * (long)kL + j] + v[(b * kH + g) * (long)kL + i];
        nv[g] = s;
    }
#pragma unroll
    for (int g = 0; g < kH; ++g) base[(long)g * kL * kL] = nv[g];
}

// pooled[b,d] = sum_l node[b,l,d] / clip(sum_l mask_ids[b,l], 1)
__global__ void k_pool(const float* __restrict__ node, const int* __restrict__ mask_ids,
                       float* __restrict__ pooled)
{
    __shared__ float vlen;
    const int b = blockIdx.x;
    if (threadIdx.x == 0) {
        int s = 0;
        for (int l = 0; l < kL; ++l) s += mask_ids[b * kL + l];
        vlen = (float)(s < 1 ? 1 : s);
    }
    __syncthreads();
    const int d = threadIdx.x;               // blockDim == kAP
    float s = 0.0f;
    for (int l = 0; l < kL; ++l) s += node[((long)b * kL + l) * kAP + d];
    pooled[b * kAP + d] = s / vlen;
}

__global__ void k_logits(const float* __restrict__ pooled, const float* __restrict__ cls_w,
                         const float* __restrict__ cls_b, float* __restrict__ outp)
{
    const int idx = blockIdx.x * blockDim.x + threadIdx.x;
    if (idx >= kB * kP) return;
    const int b = idx / kP, p = idx % kP;
    float s = cls_b[p];
    for (int d = 0; d < kA; ++d) s += pooled[b * kAP + d] * cls_w[p * kA + d];
    outp[idx] = s;
}

// ---------------------------------------------------------------------------
extern "C" void kernel_launch(void* const* d_in, const int* in_sizes, int n_in,
                              void* d_out, int out_size, void* d_ws, size_t ws_size,
                              hipStream_t stream)
{
    (void)in_sizes; (void)n_in; (void)out_size; (void)ws_size;

    const float* seq    = (const float*)d_in[0];
    const float* syn    = (const float*)d_in[1];
    const float* ln_a   = (const float*)d_in[2];
    const float* ln_b   = (const float*)d_in[3];
    const float* Wxx_w  = (const float*)d_in[4];
    const float* Wxx_b  = (const float*)d_in[5];
    const float* q_w    = (const float*)d_in[6];
    const float* q_b    = (const float*)d_in[7];
    const float* kw_w   = (const float*)d_in[8];
    const float* kw_b   = (const float*)d_in[9];
    const float* W_w    = (const float*)d_in[10];
    const float* W_b    = (const float*)d_in[11];
    const float* Wx_w   = (const float*)d_in[12];
    const float* Wx_b   = (const float*)d_in[13];
    const float* agg_w  = (const float*)d_in[14];
    const float* agg_b  = (const float*)d_in[15];
    const float* cls_w  = (const float*)d_in[16];
    const float* cls_b  = (const float*)d_in[17];
    const int*   mask_ids = (const int*)d_in[18];
    const int*   src_mask = (const int*)d_in[19];
    float* out = (float*)d_out;

    // Bump allocator over d_ws (needs ~130 MB total).
    char* wsb = (char*)d_ws;
    size_t off = 0;
    auto alloc = [&](size_t bytes) -> void* {
        void* p = wsb + off;
        off = (off + bytes + 255) & ~(size_t)255;
        return p;
    };
    _Float16* lnh   = (_Float16*)alloc((size_t)kM * kD * 2);          // 12.6 MB
    _Float16* wxxh  = (_Float16*)alloc((size_t)kAP * kD * 2);
    _Float16* qwh   = (_Float16*)alloc((size_t)kAP * kAP * 2);
    _Float16* kwh   = (_Float16*)alloc((size_t)kAP * kAP * 2);
    _Float16* wwh   = (_Float16*)alloc((size_t)kAP * kAP * 2);
    _Float16* aggh  = (_Float16*)alloc((size_t)kAP * (3 * kAP) * 2);
    float*    x32   = (float*)alloc((size_t)kM * kAP * 4);            // 4 MB
    _Float16* xcat  = (_Float16*)alloc((size_t)kM * (3 * kAP) * 2);   // 6.3 MB
    float*    q32   = (float*)alloc((size_t)kM * kAP * 4);
    float*    k32   = (float*)alloc((size_t)kM * kAP * 4);
    _Float16* qp    = (_Float16*)alloc((size_t)kB * kH * kL * 32 * 2);
    _Float16* kp    = (_Float16*)alloc((size_t)kB * kH * kL * 32 * 2);
    float*    adj   = (float*)alloc((size_t)kB * kH * kL * kL * 4);   // 67 MB
    _Float16* adjh  = (_Float16*)alloc((size_t)kB * kL * kL * 2);     // 8.4 MB
    _Float16* xT    = (_Float16*)alloc((size_t)kB * kAP * kL * 2);    // 2 MB
    float*    ax32  = (float*)alloc((size_t)kM * kAP * 4);
    _Float16* axh   = (_Float16*)alloc((size_t)kM * kAP * 2);
    float*    xg32  = (float*)alloc((size_t)kM * kAP * 4);
    float*    node32= (float*)alloc((size_t)kM * kAP * 4);
    float*    ubuf  = (float*)alloc((size_t)kB * kH * kL * 4);
    float*    vbuf  = (float*)alloc((size_t)kB * kH * kL * 4);
    float*    pooled= (float*)alloc((size_t)kB * kAP * 4);

    auto blocks1d = [](long n, int tpb) { return dim3((unsigned)((n + tpb - 1) / tpb)); };

    // --- pack weights (zero-padded f16) ---
    k_pack_weight<<<blocks1d((long)kAP * kD, 256), 256, 0, stream>>>(
        Wxx_w, kA, kD, kD, 0, wxxh, kD, 0, kAP, kD);
    k_pack_weight<<<blocks1d((long)kAP * kAP, 256), 256, 0, stream>>>(
        q_w, kA, kA, kA, 0, qwh, kAP, 0, kAP, kAP);
    k_pack_weight<<<blocks1d((long)kAP * kAP, 256), 256, 0, stream>>>(
        kw_w, kA, kA, kA, 0, kwh, kAP, 0, kAP, kAP);
    k_pack_weight<<<blocks1d((long)kAP * kAP, 256), 256, 0, stream>>>(
        W_w, kA, kA, kA, 0, wwh, kAP, 0, kAP, kAP);
    for (int i = 0; i < kNL + 1; ++i)
        k_pack_weight<<<blocks1d((long)kAP * kAP, 256), 256, 0, stream>>>(
            agg_w, kA, kA, 3 * kA, i * kA, aggh, 3 * kAP, i * kAP, kAP, kAP);

    // --- LayerNorm -> f16 ---
    k_layernorm<<<dim3(kM), 256, 0, stream>>>(seq, ln_a, ln_b, lnh);

    // --- x = LN @ Wxx^T + b  (M=8192, N=128, K=768) ---
    k_wmma_gemm<<<dim3(kM / 16, kAP / 32, 1), 32, 0, stream>>>(
        lnh, kD, 0, wxxh, kD, 0, x32, kAP, 0, kD, Wxx_b, kA, 0);
    k_f32_to_f16<<<blocks1d((long)kM * kAP, 256), 256, 0, stream>>>(
        x32, kAP, xcat, 3 * kAP, 0, kM, kAP);   // layer_list[0]

    // --- q / k projections (K=128) ---
    k_wmma_gemm<<<dim3(kM / 16, kAP / 32, 1), 32, 0, stream>>>(
        xcat, 3 * kAP, 0, qwh, kAP, 0, q32, kAP, 0, kAP, q_b, kA, 0);
    k_wmma_gemm<<<dim3(kM / 16, kAP / 32, 1), 32, 0, stream>>>(
        xcat, 3 * kAP, 0, kwh, kAP, 0, k32, kAP, 0, kAP, kw_b, kA, 0);
    k_pack_qk<<<blocks1d((long)kB * kH * kL * 32, 256), 256, 0, stream>>>(q32, k32, qp, kp);

    // --- scores = q k^T / sqrt(DK)  (batched over B*H, K=32) ---
    k_wmma_gemm<<<dim3(kL / 16, kL / 32, kB * kH), 32, 0, stream>>>(
        qp, 32, (long)kL * 32, kp, 32, (long)kL * 32,
        adj, kL, (long)kL * kL, 32, nullptr, 0, 0);

    // --- softmax(mask + syntax) ---
    k_softmax<<<dim3(kB * kH * kL), 256, 0, stream>>>(adj, syn, src_mask);

    // --- GCN layers ---
    for (int layer = 0; layer < kNL; ++layer) {
        const float* xg_src = (layer == 0) ? x32 : xg32;
        k_adjsum<<<blocks1d((long)kB * kL * kL, 256), 256, 0, stream>>>(adj, adjh);
        k_transpose_x<<<blocks1d((long)kB * kAP * kL, 256), 256, 0, stream>>>(xg_src, xT);
        // Ax[b] = adjsum[b] @ x_g[b]  (M=512, N=128, K=512, batch=B)
        k_wmma_gemm<<<dim3(kL / 16, kAP / 32, kB), 32, 0, stream>>>(
            adjh, kL, (long)kL * kL, xT, kL, (long)kAP * kL,
            ax32, kAP, (long)kL * kAP, kL, nullptr, 0, 0);
        k_f32_to_f16<<<blocks1d((long)kM * kAP, 256), 256, 0, stream>>>(
            ax32, kAP, axh, kAP, 0, kM, kAP);
        // x_g = relu(Ax @ W^T + b)
        k_wmma_gemm<<<dim3(kM / 16, kAP / 32, 1), 32, 0, stream>>>(
            axh, kAP, 0, wwh, kAP, 0, xg32, kAP, 0, kAP, W_b, kA, 1);
        k_f32_to_f16<<<blocks1d((long)kM * kAP, 256), 256, 0, stream>>>(
            xg32, kAP, xcat, 3 * kAP, (long)(layer + 1) * kAP, kM, kAP);
        // edge update terms
        k_uv<<<blocks1d((long)kB * kH * kL, 256), 256, 0, stream>>>(xg32, Wx_w, ubuf, vbuf);
        k_adj_update<<<blocks1d((long)kB * kL * kL, 256), 256, 0, stream>>>(
            adj, ubuf, vbuf, Wx_w, Wx_b);
    }

    // --- node = relu(concat(layers) @ agg^T + b)  (K=384) ---
    k_wmma_gemm<<<dim3(kM / 16, kAP / 32, 1), 32, 0, stream>>>(
        xcat, 3 * kAP, 0, aggh, 3 * kAP, 0, node32, kAP, 0, 3 * kAP, agg_b, kA, 1);

    // --- masked mean pool + classifier ---
    k_pool<<<dim3(kB), kAP, 0, stream>>>(node32, mask_ids, pooled);
    k_logits<<<dim3(1), 64, 0, stream>>>(pooled, cls_w, cls_b, out);
}
